// GaussianLSS_55027120996692
// MI455X (gfx1250) — matrix operand order repaired and checked by reading, not compiled, not run on hardware
//
#include <hip/hip_runtime.h>
#include <hip/hip_bf16.h>

// ---------------- static config (mirrors reference) ----------------
#define NCAM   6
#define FC     128
#define FH     16
#define FW     44
#define PIX    (FH*FW)          // 704
#define PH     (FH+2)           // 18 padded
#define PW     (FW+2)           // 46 padded
#define PPIX   (PH*PW)          // 828
#define ND     48               // depth bins
#define OUTC   128
#define COUT3  (OUTC+ND+1)      // 177
#define GTOT   (NCAM*PIX)       // 4224 gaussians
#define NCHUNK (GTOT/32)        // 132
#define BEVH   100
#define BEVW   100
#define BEVP   (BEVH*BEVW)      // 10000 pixels

#if __has_builtin(__builtin_amdgcn_global_load_async_to_lds_b128) && \
    __has_builtin(__builtin_amdgcn_s_wait_asynccnt)
#define HAVE_ASYNC_LDS 1
#else
#define HAVE_ASYNC_LDS 0
#endif

typedef __attribute__((ext_vector_type(16))) _Float16 v16h;
typedef __attribute__((ext_vector_type(8)))  _Float16 h8;
typedef __attribute__((ext_vector_type(8)))  float    v8f;
typedef int v4i __attribute__((vector_size(16)));   // matches async builtin param
#if HAVE_ASYNC_LDS
typedef __attribute__((address_space(1))) v4i* gptr_v4i;
typedef __attribute__((address_space(3))) v4i* lptr_v4i;
#endif

// ---- WMMA fragment helpers (16-bit A 16x32 / B 32x16, wave32) ----
// A: lane l holds row M=l%16; element e -> K = (e/8)*16 + (l/16)*8 + (e%8)
//    => two contiguous 8-half runs at K = hl*8 and K = 16+hl*8.
// B: lane l holds col N=l%16; element e -> K = (e/8)*16 + (l/16)*8 + (e%8)
//    => two contiguous 8-half runs at K = hl*8 and K = 16+hl*8.
__device__ inline v16h a_frag(const _Float16* row, int hl) {
    h8 lo = *(const h8*)(row + hl * 8);
    h8 hi = *(const h8*)(row + 16 + hl * 8);
    v16h a;
#pragma unroll
    for (int e = 0; e < 8; ++e) { a[e] = lo[e]; a[e + 8] = hi[e]; }
    return a;
}
__device__ inline v16h b_frag(const _Float16* krow, int hl) {
    h8 lo = *(const h8*)(krow + hl * 8);
    h8 hi = *(const h8*)(krow + 16 + hl * 8);
    v16h b;
#pragma unroll
    for (int e = 0; e < 8; ++e) { b[e] = lo[e]; b[e + 8] = hi[e]; }
    return b;
}

__device__ inline void inv3(const float* m, float* o) {
    float a=m[0],b=m[1],c=m[2],d=m[3],e=m[4],f=m[5],g=m[6],h=m[7],i=m[8];
    float A=e*i-f*h, B=c*h-b*i, C=b*f-c*e;
    float D=f*g-d*i, E=a*i-c*g, F=c*d-a*f;
    float G=d*h-e*g, H=b*g-a*h, I=a*e-b*d;
    float inv = 1.f/(a*A + b*D + c*G);
    o[0]=A*inv; o[1]=B*inv; o[2]=C*inv;
    o[3]=D*inv; o[4]=E*inv; o[5]=F*inv;
    o[6]=G*inv; o[7]=H*inv; o[8]=I*inv;
}

// ---------------- 1. geometry lifting (scalar VALU, tiny) ----------------
__global__ void k_geometry(const float* __restrict__ rot, const float* __restrict__ trans,
                           const float* __restrict__ intr, const float* __restrict__ prot,
                           const float* __restrict__ ptrans, float* __restrict__ geo) {
    int idx = blockIdx.x * 256 + threadIdx.x;
    if (idx >= NCAM * ND * PIX) return;
    int p = idx % PIX; int t = idx / PIX; int d = t % ND; int n = t / ND;
    int h = p / FW, w = p % FW;
    float fx = (float)w * (351.f / 43.f);
    float fy = (float)h * (127.f / 15.f);
    float fz = 1.f + (float)d;
    float iP[9]; inv3(prot + n * 9, iP);
    float sx = fx - ptrans[n*3+0], sy = fy - ptrans[n*3+1], sz = fz - ptrans[n*3+2];
    float vx = iP[0]*sx + iP[1]*sy + iP[2]*sz;
    float vy = iP[3]*sx + iP[4]*sy + iP[5]*sz;
    float vz = iP[6]*sx + iP[7]*sy + iP[8]*sz;
    float qx = vx * vz, qy = vy * vz, qz = vz;
    float iK[9]; inv3(intr + n * 9, iK);
    const float* R = rot + n * 9;
    float cmb[9];
#pragma unroll
    for (int r = 0; r < 3; ++r)
#pragma unroll
        for (int c = 0; c < 3; ++c)
            cmb[r*3+c] = R[r*3+0]*iK[c] + R[r*3+1]*iK[3+c] + R[r*3+2]*iK[6+c];
    geo[idx*3+0] = cmb[0]*qx + cmb[1]*qy + cmb[2]*qz + trans[n*3+0];
    geo[idx*3+1] = cmb[3]*qx + cmb[4]*qy + cmb[5]*qz + trans[n*3+1];
    geo[idx*3+2] = cmb[6]*qx + cmb[7]*qy + cmb[8]*qz + trans[n*3+2];
}

// ---------------- conversion / layout kernels ----------------
// NCHW f32 -> zero-padded NHWC f16: [n][PH][PW][128]
__global__ void k_cvt_pad(const float* __restrict__ s, _Float16* __restrict__ d) {
    int i = blockIdx.x * 256 + threadIdx.x;
    if (i >= NCAM * PPIX * FC) return;
    int c = i & 127; int t = i >> 7;
    int ww = t % PW; t /= PW;
    int hh = t % PH; int n = t / PH;
    int h = hh - 1, w = ww - 1;
    bool inb = ((unsigned)h < (unsigned)FH) & ((unsigned)w < (unsigned)FW);
    d[i] = inb ? (_Float16)s[((size_t)n * FC + c) * PIX + h * FW + w] : (_Float16)0.f;
}

__global__ void k_zero_u32(unsigned* __restrict__ p, int n) {
    int i = blockIdx.x * 256 + threadIdx.x;
    if (i < n) p[i] = 0u;
}

// conv 3x3 weights: [oc][ic][kh][kw] f32 -> [oc][(kh*3+kw)*128 + ic] f16
__global__ void k_cvt_w33(const float* __restrict__ s, _Float16* __restrict__ d) {
    int i = blockIdx.x * 256 + threadIdx.x;
    if (i >= FC * FC * 9) return;
    int oc = i / (FC * 9); int rem = i % (FC * 9);
    int pos = rem / FC; int ic = rem % FC;
    d[i] = (_Float16)s[oc * (FC * 9) + ic * 9 + pos];
}

// conv3 1x1 weights: [177][128] -> [192][128] f16 zero-padded rows
__global__ void k_cvt_w11(const float* __restrict__ s, _Float16* __restrict__ d) {
    int i = blockIdx.x * 256 + threadIdx.x;
    if (i >= 192 * FC) return;
    int oc = i / FC, ic = i % FC;
    d[i] = (oc < COUT3) ? (_Float16)s[oc * FC + ic] : (_Float16)0.f;
}

// fold conv bias + eval BN into y = x*sc + sh
__global__ void k_bnfold(const float* __restrict__ b, const float* __restrict__ g,
                         const float* __restrict__ be, const float* __restrict__ m,
                         const float* __restrict__ v, float* __restrict__ sc,
                         float* __restrict__ sh) {
    int i = threadIdx.x;
    if (i < FC) {
        float s = g[i] * rsqrtf(v[i] + 1e-3f);
        sc[i] = s;
        sh[i] = (b[i] - m[i]) * s + be[i];
    }
}

// ---------------- 2. depth-net 3x3 conv: implicit GEMM, WMMA -------------
// Padded-NHWC in/out.  One wave per (image, oc-tile, pixel-tile).
// Fully unrolled K loop: every load is [base + constant-imm] b128.
__global__ __launch_bounds__(256)
void k_conv3x3(const _Float16* __restrict__ xp, const _Float16* __restrict__ wh,
               const float* __restrict__ sc, const float* __restrict__ sh,
               _Float16* __restrict__ yp) {
    int wave = blockIdx.x * 8 + (threadIdx.x >> 5);
    int lane = threadIdx.x & 31;
    int nt = wave % (PIX / 16); int t = wave / (PIX / 16);
    int mt = t % 8; int n = t / 8;
    int hl = lane >> 4, l16 = lane & 15;
    int p  = nt * 16 + l16;                 // B/N pixel for this lane
    int ph = p / FW, pw = p % FW;
    const _Float16* wbase = wh + (size_t)(mt * 16 + l16) * (9 * FC);
    const _Float16* xpix  = xp + ((size_t)n * PPIX + (size_t)ph * PW + pw) * FC;
    v8f acc = {};
#pragma unroll
    for (int pos = 0; pos < 9; ++pos) {     // kernel tap; offsets all constant
        const _Float16* krowb = xpix + ((pos / 3) * PW + (pos % 3)) * FC;
#pragma unroll
        for (int q = 0; q < 4; ++q) {       // 32-wide K slices of 128 ic
            v16h a = a_frag(wbase + (pos * 4 + q) * 32, hl);
            v16h b = b_frag(krowb + q * 32, hl);
            acc = __builtin_amdgcn_wmma_f32_16x16x32_f16(false, a, false, b,
                                                         (short)0, acc, false, false);
        }
    }
    // epilogue: y = relu(x*sc + sh); one b128 store (8 consecutive oc / lane)
    int ocb = mt * 16 + hl * 8;             // D: M = (lane/16)*8 + r
    float4 s0 = *(const float4*)(sc + ocb), s1 = *(const float4*)(sc + ocb + 4);
    float4 h0 = *(const float4*)(sh + ocb), h1 = *(const float4*)(sh + ocb + 4);
    float scv[8] = {s0.x, s0.y, s0.z, s0.w, s1.x, s1.y, s1.z, s1.w};
    float shv[8] = {h0.x, h0.y, h0.z, h0.w, h1.x, h1.y, h1.z, h1.w};
    size_t obase = ((size_t)n * PPIX + (size_t)(ph + 1) * PW + (pw + 1)) * FC + ocb;
    h8 ov;
#pragma unroll
    for (int r = 0; r < 8; ++r)
        ov[r] = (_Float16)fmaxf(acc[r] * scv[r] + shv[r], 0.f);
    *(h8*)(yp + obase) = ov;
}

// ---------------- 2b. 1x1 conv head (WMMA), NHWC f32 out -----------------
__global__ __launch_bounds__(256)
void k_conv1x1(const _Float16* __restrict__ xp, const _Float16* __restrict__ wh,
               const float* __restrict__ bias, float* __restrict__ y) {
    int wave = blockIdx.x * 8 + (threadIdx.x >> 5);
    int lane = threadIdx.x & 31;
    int nt = wave % (PIX / 16); int t = wave / (PIX / 16);
    int mt = t % 12; int n = t / 12;
    int hl = lane >> 4, l16 = lane & 15;
    int p = nt * 16 + l16;
    int ph = p / FW, pw = p % FW;
    const _Float16* wbase = wh + (size_t)(mt * 16 + l16) * FC;
    const _Float16* krow0 = xp + ((size_t)n * PPIX + (size_t)(ph + 1) * PW + (pw + 1)) * FC;
    v8f acc = {};
#pragma unroll
    for (int kc = 0; kc < 4; ++kc) {
        v16h a = a_frag(wbase + kc * 32, hl);
        v16h b = b_frag(krow0 + kc * 32, hl);
        acc = __builtin_amdgcn_wmma_f32_16x16x32_f16(false, a, false, b,
                                                     (short)0, acc, false, false);
    }
    size_t obase = (size_t)(n * PIX + p) * COUT3;
#pragma unroll
    for (int r = 0; r < 8; ++r) {
        int oc = mt * 16 + hl * 8 + r;
        if (oc < COUT3) y[obase + oc] = acc[r] + bias[oc];
    }
}

// ---------------- 3. per-gaussian depth moments -> splat params ----------
// x3 is NHWC: [n][p][177]; logits contiguous per pixel.
__global__ void k_gauss(const float* __restrict__ x3, const float* __restrict__ geo,
                        float* __restrict__ gp) {
    int g = blockIdx.x * 256 + threadIdx.x;
    if (g >= GTOT) return;
    int n = g / PIX, p = g % PIX;
    const float* lg = x3 + (size_t)g * COUT3;
    float mx = -1e30f;
#pragma unroll 8
    for (int d = 0; d < ND; ++d) mx = fmaxf(mx, lg[d]);
    float s = 0.f;
#pragma unroll 8
    for (int d = 0; d < ND; ++d) s += __expf(lg[d] - mx);
    float inv = 1.f / s;
    float m0 = 0.f, m1 = 0.f, sxx = 0.f, sxy = 0.f, syy = 0.f;
#pragma unroll 4
    for (int d = 0; d < ND; ++d) {
        float pr = __expf(lg[d] - mx) * inv;
        const float* q = geo + ((size_t)(n * ND + d) * PIX + p) * 3;
        float gx = q[0], gy = q[1];
        m0 += pr * gx; m1 += pr * gy;
        sxx += pr * gx * gx; sxy += pr * gx * gy; syy += pr * gy * gy;
    }
    const float k9 = 1.f / 9.f;                 // ERR_TOL^2 / 9
    float cxx = (sxx - m0 * m0) * k9;
    float cxy = (sxy - m0 * m1) * k9;
    float cyy = (syy - m1 * m1) * k9;
    float mnx = m0 * 0.02f, mny = m1 * 0.02f;   // normalize to [-1,1]
    float a = 2500.f * (cyy * 4e-4f) + 0.3f;    // sh^2*cov_n_yy + dilate
    float b = 2500.f * (cxy * 4e-4f);
    float c = 2500.f * (cxx * 4e-4f) + 0.3f;
    float det = a * c - b * b;
    float op = 1.f / (1.f + __expf(-lg[ND]));
    op = op > 0.05f ? op : 0.f;
    float invd = 0.f;
    if (det > 0.f) invd = 1.f / det; else op = 0.f;
    float* o = gp + (size_t)g * 8;
    o[0] = 50.f - 50.f * mny;                   // py
    o[1] = 50.f - 50.f * mnx;                   // px
    o[2] = -0.5f * c * invd;                    // dy^2 coef
    o[3] = -0.5f * a * invd;                    // dx^2 coef
    o[4] = b * invd;                            // dy*dx coef
    o[5] = op;
    o[6] = 0.f; o[7] = 0.f;
}

// splat features, transposed: fthT[c][g] f16 (conv3 channels ND+1..ND+128)
__global__ void k_packT(const float* __restrict__ x3, _Float16* __restrict__ f) {
    int i = blockIdx.x * 256 + threadIdx.x;
    if (i >= OUTC * GTOT) return;
    int c = i / GTOT, g = i % GTOT;
    f[i] = (_Float16)x3[(size_t)g * COUT3 + ND + 1 + c];
}

// ---------------- 4. BEV splatting: transmittance scan + WMMA GEMM -------
// 32 pixels x 128 channels per block (128 thr = 4 waves).
// A = featT (M=channel, K=g) from global (two b128 / frag, prefetched).
// B = wgt   (N=pixel,   K=g) from LDS   (two b128 / frag).
// Gaussian params double-buffered into LDS with async-to-LDS when available.
__global__ __launch_bounds__(128)
void k_raster(const float* __restrict__ gp, const _Float16* __restrict__ fthT,
              float* __restrict__ bev) {
    __shared__ float    alpha_lds[32 * 32];
    __shared__ __align__(16) _Float16 wgt_lds[32 * 32];
    __shared__ float    Tl[32];
    __shared__ __align__(16) float gplds[2][32 * 8];
    int tid = threadIdx.x, lane = tid & 31, wid = tid >> 5;
    int hl = lane >> 4, l16 = lane & 15;
    int pbase = blockIdx.x * 32;
    if (tid < 32) Tl[tid] = 1.f;
    int c0 = wid * 32, c1 = c0 + 16;
    const _Float16* arow0 = fthT + (size_t)(c0 + l16) * GTOT;
    const _Float16* arow1 = fthT + (size_t)(c1 + l16) * GTOT;
    v8f acc00 = {}, acc01 = {}, acc10 = {}, acc11 = {};
#if HAVE_ASYNC_LDS
    if (tid < 64)   // prime chunk 0: 64 lanes x 16B = 1KB of gaussian params
        __builtin_amdgcn_global_load_async_to_lds_b128(
            (gptr_v4i)((const float4*)gp + tid),
            (lptr_v4i)((float4*)gplds[0] + tid), 0, 0);
#endif
    int buf = 0;
    for (int kc = 0; kc < NCHUNK; ++kc) {
        int g0 = kc * 32;
#if HAVE_ASYNC_LDS
        if (tid < 64) __builtin_amdgcn_s_wait_asynccnt(0);
        __syncthreads();                 // buf arrived + prior-iter reads done
        if (tid < 64 && kc + 1 < NCHUNK)  // overlap next chunk with this one
            __builtin_amdgcn_global_load_async_to_lds_b128(
                (gptr_v4i)((const float4*)(gp + (size_t)(g0 + 32) * 8) + tid),
                (lptr_v4i)((float4*)gplds[buf ^ 1] + tid), 0, 0);
#else
        __syncthreads();
        if (tid < 64) ((float4*)gplds[buf])[tid] = *((const float4*)(gp + (size_t)g0 * 8) + tid);
        __syncthreads();
#endif
        if (g0 + 64 < GTOT) {            // prefetch A rows two chunks ahead
            __builtin_prefetch(arow0 + g0 + 64, 0, 3);
            __builtin_prefetch(arow1 + g0 + 64, 0, 3);
        }
        // alpha tile 32x32 = 1024 entries, 8 per thread (exp-heavy, parallel)
#pragma unroll
        for (int i = 0; i < 8; ++i) {
            int idx = tid + i * 128;
            int p = idx >> 5, gg = idx & 31;
            const float* G = gplds[buf] + gg * 8;
            float4 ga = *(const float4*)G;
            float4 gb = *(const float4*)(G + 4);
            int P = pbase + p;
            float dy = ga.x - (float)(P / BEVW);
            float dx = ga.y - (float)(P % BEVW);
            float pw = fminf(ga.z * dy * dy + ga.w * dx * dx + gb.x * dy * dx, 0.f);
            float al = fminf(0.99f, gb.y * __expf(pw));
            al = (al < (1.f / 255.f)) ? 0.f : al;
            if (P >= BEVP) al = 0.f;     // tail tile guard
            alpha_lds[idx] = al;
        }
        __syncthreads();
        if (tid < 32) {                  // sequential transmittance per pixel
            float t = Tl[tid];
#pragma unroll
            for (int gg = 0; gg < 32; ++gg) {
                float al = alpha_lds[tid * 32 + gg];
                wgt_lds[tid * 32 + gg] = (_Float16)(al * t);
                t *= (1.f - al);
            }
            Tl[tid] = t;
        }
        __syncthreads();
        v16h b0 = b_frag(&wgt_lds[l16 * 32], hl);
        v16h b1 = b_frag(&wgt_lds[(16 + l16) * 32], hl);
        v16h a0 = a_frag(arow0 + g0, hl);
        v16h a1 = a_frag(arow1 + g0, hl);
        acc00 = __builtin_amdgcn_wmma_f32_16x16x32_f16(false, a0, false, b0,
                                                       (short)0, acc00, false, false);
        acc01 = __builtin_amdgcn_wmma_f32_16x16x32_f16(false, a0, false, b1,
                                                       (short)0, acc01, false, false);
        acc10 = __builtin_amdgcn_wmma_f32_16x16x32_f16(false, a1, false, b0,
                                                       (short)0, acc10, false, false);
        acc11 = __builtin_amdgcn_wmma_f32_16x16x32_f16(false, a1, false, b1,
                                                       (short)0, acc11, false, false);
        buf ^= 1;
    }
    // D: M = channel = (lane/16)*8 + r ; N = pixel = lane%16 (+16 for b1 tiles)
#pragma unroll
    for (int r = 0; r < 8; ++r) {
        int ch0 = c0 + hl * 8 + r, ch1 = c1 + hl * 8 + r;
        int P0 = pbase + l16, P1 = pbase + 16 + l16;
        if (P0 < BEVP) {
            bev[(size_t)ch0 * BEVP + P0] = acc00[r];
            bev[(size_t)ch1 * BEVP + P0] = acc10[r];
        }
        if (P1 < BEVP) {
            bev[(size_t)ch0 * BEVP + P1] = acc01[r];
            bev[(size_t)ch1 * BEVP + P1] = acc11[r];
        }
    }
}

// ---------------- host orchestration ----------------
extern "C" void kernel_launch(void* const* d_in, const int* in_sizes, int n_in,
                              void* d_out, int out_size, void* d_ws, size_t ws_size,
                              hipStream_t stream) {
    (void)in_sizes; (void)n_in; (void)out_size; (void)ws_size;
    const float* rot    = (const float*)d_in[0];
    const float* trans  = (const float*)d_in[1];
    const float* intr   = (const float*)d_in[2];
    const float* prot   = (const float*)d_in[3];
    const float* ptrans = (const float*)d_in[4];
    const float* imgf   = (const float*)d_in[5];
    const float* w1 = (const float*)d_in[6];  const float* b1 = (const float*)d_in[7];
    const float* g1 = (const float*)d_in[8];  const float* be1 = (const float*)d_in[9];
    const float* m1 = (const float*)d_in[10]; const float* v1 = (const float*)d_in[11];
    const float* w2 = (const float*)d_in[12]; const float* b2 = (const float*)d_in[13];
    const float* g2 = (const float*)d_in[14]; const float* be2 = (const float*)d_in[15];
    const float* m2 = (const float*)d_in[16]; const float* v2 = (const float*)d_in[17];
    const float* w3 = (const float*)d_in[18]; const float* b3 = (const float*)d_in[19];

    char* ws = (char*)d_ws;
    size_t off = 0;
    auto take = [&](size_t bytes) -> char* {
        char* p = ws + off;
        off += (bytes + 255) & ~(size_t)255;
        return p;
    };
    const size_t NPAD = (size_t)NCAM * PPIX * FC;           // 635904
    float*    geo  = (float*)   take((size_t)NCAM * ND * PIX * 3 * 4);
    _Float16* x0p  = (_Float16*)take(NPAD * 2);
    _Float16* x1p  = (_Float16*)take(NPAD * 2);
    _Float16* x2p  = (_Float16*)take(NPAD * 2);
    _Float16* w1h  = (_Float16*)take((size_t)FC * FC * 9 * 2);
    _Float16* w2h  = (_Float16*)take((size_t)FC * FC * 9 * 2);
    _Float16* w3h  = (_Float16*)take((size_t)192 * FC * 2);
    float*    x3f  = (float*)   take((size_t)NCAM * PIX * COUT3 * 4);
    float*    gpar = (float*)   take((size_t)GTOT * 8 * 4);
    _Float16* fthT = (_Float16*)take((size_t)OUTC * GTOT * 2);
    float*    sc1  = (float*)take(FC * 4);  float* sh1 = (float*)take(FC * 4);
    float*    sc2  = (float*)take(FC * 4);  float* sh2 = (float*)take(FC * 4);

    const int NGEO = NCAM * ND * PIX;
    k_geometry<<<(NGEO + 255) / 256, 256, 0, stream>>>(rot, trans, intr, prot, ptrans, geo);
    k_cvt_pad<<<((int)NPAD + 255) / 256, 256, 0, stream>>>(imgf, x0p);
    k_zero_u32<<<((int)(NPAD / 2) + 255) / 256, 256, 0, stream>>>((unsigned*)x1p, (int)(NPAD / 2));
    k_cvt_w33<<<(FC * FC * 9 + 255) / 256, 256, 0, stream>>>(w1, w1h);
    k_cvt_w33<<<(FC * FC * 9 + 255) / 256, 256, 0, stream>>>(w2, w2h);
    k_cvt_w11<<<(192 * FC + 255) / 256, 256, 0, stream>>>(w3, w3h);
    k_bnfold<<<1, 128, 0, stream>>>(b1, g1, be1, m1, v1, sc1, sh1);
    k_bnfold<<<1, 128, 0, stream>>>(b2, g2, be2, m2, v2, sc2, sh2);

    // waves = 6 images * 8 oc-tiles * 44 pixel-tiles = 2112 -> 264 blocks
    k_conv3x3<<<264, 256, 0, stream>>>(x0p, w1h, sc1, sh1, x1p);
    k_conv3x3<<<264, 256, 0, stream>>>(x1p, w2h, sc2, sh2, x2p);
    // waves = 6 * 12 * 44 = 3168 -> 396 blocks
    k_conv1x1<<<396, 256, 0, stream>>>(x2p, w3h, b3, x3f);

    k_gauss<<<(GTOT + 255) / 256, 256, 0, stream>>>(x3f, geo, gpar);
    k_packT<<<(OUTC * GTOT + 255) / 256, 256, 0, stream>>>(x3f, fthT);

    // 10000 pixels / 32 per block = 313 blocks, 128 threads (4 waves)
    k_raster<<<(BEVP + 31) / 32, 128, 0, stream>>>(gpar, fthT, (float*)d_out);
}
